// _RWKV6BlockSimple_52450140619470
// MI455X (gfx1250) — matrix operand order repaired and checked by reading, not compiled, not run on hardware
//
#include <hip/hip_runtime.h>
#include <hip/hip_bf16.h>

// ---------------------------------------------------------------------------
// RWKV block for MI455X (gfx1250, wave32, WMMA).
// B=4, T=2048, C=2048.  GEMMs in bf16 WMMA (16x16x32, f32 accum) with
// double-buffered LDS tiles and GLOBAL_LOAD_ASYNC_TO_LDS_B128 staging.
// LN / mix / scan in fp32 streaming kernels.
// ---------------------------------------------------------------------------

#define Cdim 2048
#define Tdim 2048
#define Bdim 4
#define Mrows (Bdim * Tdim)            // 8192
#define NELEM ((size_t)Mrows * Cdim)   // 16,777,216

typedef __attribute__((ext_vector_type(16))) __bf16 v16bf;
typedef __attribute__((ext_vector_type(8)))  float  v8f;
typedef int v4i_ld __attribute__((vector_size(16)));

union FragA { unsigned int u[8]; v16bf v; };

#if defined(__gfx1250__) && __has_builtin(__builtin_amdgcn_global_load_async_to_lds_b128) && \
    __has_builtin(__builtin_amdgcn_s_wait_asynccnt)
#define USE_ASYNC_LDS 1
// memory -> LDS direct copy, tracked by ASYNCcnt (16B per lane)
#define GL_ASYNC_B128(gptr, lptr)                                        \
  __builtin_amdgcn_global_load_async_to_lds_b128(                        \
      (__attribute__((address_space(1))) v4i_ld*)(gptr),                 \
      (__attribute__((address_space(3))) v4i_ld*)(lptr), 0, 0)
#else
#define USE_ASYNC_LDS 0
#endif

// ---- bf16 helpers ----------------------------------------------------------
__device__ __forceinline__ unsigned short f2bf(float f) {
  unsigned int u = __float_as_uint(f);
  if ((u & 0x7fffffffu) > 0x7f800000u) return (unsigned short)((u >> 16) | 0x40); // NaN
  unsigned int r = (u + 0x7fffu + ((u >> 16) & 1u)) >> 16;
  return (unsigned short)r;
}
__device__ __forceinline__ unsigned int pack_bf2(float lo, float hi) {
  return (unsigned int)f2bf(lo) | ((unsigned int)f2bf(hi) << 16);
}
__device__ __forceinline__ float sigmoidf(float x) {
  return 1.0f / (1.0f + __expf(-x));
}

// ---------------------------------------------------------------------------
// 1) fp32 -> bf16 weight conversion (4M elements per matrix)
// ---------------------------------------------------------------------------
__global__ void wconv_kernel(const float* __restrict__ W, unsigned int* __restrict__ O) {
  int i = blockIdx.x * blockDim.x + threadIdx.x;  // handles 4 floats -> 2 uints
  const float4 v = ((const float4*)W)[i];
  O[2 * i + 0] = pack_bf2(v.x, v.y);
  O[2 * i + 1] = pack_bf2(v.z, v.w);
}

// ---------------------------------------------------------------------------
// 2) LayerNorm: one row (C=2048) per 256-thread block, wave32 reductions
// ---------------------------------------------------------------------------
__global__ void ln_kernel(const float* __restrict__ x, const float* __restrict__ lw,
                          const float* __restrict__ lb, float* __restrict__ xt) {
  const int row = blockIdx.x;
  const float4* xr4 = (const float4*)(x + (size_t)row * Cdim);
  float4 v0 = xr4[threadIdx.x * 2 + 0];
  float4 v1 = xr4[threadIdx.x * 2 + 1];
  float s  = v0.x + v0.y + v0.z + v0.w + v1.x + v1.y + v1.z + v1.w;
  float s2 = v0.x*v0.x + v0.y*v0.y + v0.z*v0.z + v0.w*v0.w
           + v1.x*v1.x + v1.y*v1.y + v1.z*v1.z + v1.w*v1.w;
  for (int o = 16; o > 0; o >>= 1) {         // wave32 reduce
    s  += __shfl_xor(s,  o);
    s2 += __shfl_xor(s2, o);
  }
  __shared__ float sh[16];
  const int wid = threadIdx.x >> 5;
  if ((threadIdx.x & 31) == 0) { sh[wid] = s; sh[8 + wid] = s2; }
  __syncthreads();
  float ts = 0.f, ts2 = 0.f;
  for (int i = 0; i < 8; ++i) { ts += sh[i]; ts2 += sh[8 + i]; }
  const float mu  = ts * (1.0f / Cdim);
  const float var = ts2 * (1.0f / Cdim) - mu * mu;
  const float rs  = rsqrtf(var + 1e-5f);

  const float4 w0 = ((const float4*)lw)[threadIdx.x * 2 + 0];
  const float4 w1 = ((const float4*)lw)[threadIdx.x * 2 + 1];
  const float4 b0 = ((const float4*)lb)[threadIdx.x * 2 + 0];
  const float4 b1 = ((const float4*)lb)[threadIdx.x * 2 + 1];
  float4 o0, o1;
  o0.x = (v0.x - mu) * rs * w0.x + b0.x;  o0.y = (v0.y - mu) * rs * w0.y + b0.y;
  o0.z = (v0.z - mu) * rs * w0.z + b0.z;  o0.w = (v0.w - mu) * rs * w0.w + b0.w;
  o1.x = (v1.x - mu) * rs * w1.x + b1.x;  o1.y = (v1.y - mu) * rs * w1.y + b1.y;
  o1.z = (v1.z - mu) * rs * w1.z + b1.z;  o1.w = (v1.w - mu) * rs * w1.w + b1.w;
  float4* ot4 = (float4*)(xt + (size_t)row * Cdim);
  ot4[threadIdx.x * 2 + 0] = o0;
  ot4[threadIdx.x * 2 + 1] = o1;
}

// ---------------------------------------------------------------------------
// 3) time-shift mix -> bf16 staging for the three projections
// ---------------------------------------------------------------------------
__global__ void mix_kernel(const float* __restrict__ xt,
                           const float* __restrict__ mk, const float* __restrict__ mv,
                           const float* __restrict__ mr,
                           unsigned int* __restrict__ xk, unsigned int* __restrict__ xv,
                           unsigned int* __restrict__ xr) {
  const int i = blockIdx.x * blockDim.x + threadIdx.x;     // float4 index
  const size_t e = (size_t)i * 4;
  const int c   = (int)(e & (Cdim - 1));
  const int row = (int)(e >> 11);
  const int t   = row & (Tdim - 1);

  const float4 cur = ((const float4*)xt)[i];
  float4 prev = make_float4(0.f, 0.f, 0.f, 0.f);
  if (t > 0) prev = ((const float4*)xt)[i - (Cdim / 4)];

  const float4 k4 = ((const float4*)mk)[c >> 2];
  const float4 v4 = ((const float4*)mv)[c >> 2];
  const float4 r4 = ((const float4*)mr)[c >> 2];

  float kx0 = cur.x * k4.x + prev.x * (1.f - k4.x);
  float kx1 = cur.y * k4.y + prev.y * (1.f - k4.y);
  float kx2 = cur.z * k4.z + prev.z * (1.f - k4.z);
  float kx3 = cur.w * k4.w + prev.w * (1.f - k4.w);
  float vx0 = cur.x * v4.x + prev.x * (1.f - v4.x);
  float vx1 = cur.y * v4.y + prev.y * (1.f - v4.y);
  float vx2 = cur.z * v4.z + prev.z * (1.f - v4.z);
  float vx3 = cur.w * v4.w + prev.w * (1.f - v4.w);
  float rx0 = cur.x * r4.x + prev.x * (1.f - r4.x);
  float rx1 = cur.y * r4.y + prev.y * (1.f - r4.y);
  float rx2 = cur.z * r4.z + prev.z * (1.f - r4.z);
  float rx3 = cur.w * r4.w + prev.w * (1.f - r4.w);

  xk[2 * i + 0] = pack_bf2(kx0, kx1);  xk[2 * i + 1] = pack_bf2(kx2, kx3);
  xv[2 * i + 0] = pack_bf2(vx0, vx1);  xv[2 * i + 1] = pack_bf2(vx2, vx3);
  xr[2 * i + 0] = pack_bf2(rx0, rx1);  xr[2 * i + 1] = pack_bf2(rx2, rx3);
}

// ---------------------------------------------------------------------------
// 4) bf16 WMMA GEMM:  O[m,n] = sum_k A[m,k] * W[n,k]   (M=8192, N=K=2048)
//    128x128 block tile, BK=64, double-buffered LDS, 8 waves each own 64x32
//    (4x2 WMMA tiles, 16 WMMAs per wave per K round).
//    gridDim.z selects (A,W,O) triple; sigmoid applied when z == sigmoid_z.
// ---------------------------------------------------------------------------
#define LSTRIDE 36   // uints per row: 32 data (64 bf16) + 4 pad -> 144B, 16B aligned

__device__ __forceinline__ v16bf load_afrag(const unsigned int* As, int tile, int lane, int h2) {
  const int m = tile * 16 + (lane & 15);
  const int half = lane >> 4;
  const unsigned int* base = As + m * LSTRIDE + h2 * 16;
  const uint4 a = *(const uint4*)(base + half * 4);       // K = half*8 ..+7
  const uint4 b = *(const uint4*)(base + 8 + half * 4);   // K = 16+half*8 ..+7
  FragA f;
  f.u[0] = a.x; f.u[1] = a.y; f.u[2] = a.z; f.u[3] = a.w;
  f.u[4] = b.x; f.u[5] = b.y; f.u[6] = b.z; f.u[7] = b.w;
  return f.v;
}
__device__ __forceinline__ v16bf load_bfrag(const unsigned int* Ws, int tile, int lane, int h2) {
  const int n = tile * 16 + (lane & 15);
  const int half = lane >> 4;
  const unsigned int* base = Ws + n * LSTRIDE + h2 * 16;
  const uint4 a = *(const uint4*)(base + half * 4);       // K = half*8 ..+7
  const uint4 b = *(const uint4*)(base + 8 + half * 4);   // K = 16+half*8 ..+7
  FragA f;
  f.u[0] = a.x; f.u[1] = a.y; f.u[2] = a.z; f.u[3] = a.w;
  f.u[4] = b.x; f.u[5] = b.y; f.u[6] = b.z; f.u[7] = b.w;
  return f.v;
}

__global__ __launch_bounds__(256)
void gemm_bf16_kernel(const unsigned short* __restrict__ A0, const unsigned short* __restrict__ A1,
                      const unsigned short* __restrict__ A2,
                      const unsigned short* __restrict__ W0, const unsigned short* __restrict__ W1,
                      const unsigned short* __restrict__ W2,
                      float* __restrict__ O0, float* __restrict__ O1, float* __restrict__ O2,
                      int sigmoid_z) {
  const unsigned short* A;
  const unsigned short* W;
  float* O;
  if (blockIdx.z == 0)      { A = A0; W = W0; O = O0; }
  else if (blockIdx.z == 1) { A = A1; W = W1; O = O1; }
  else                      { A = A2; W = W2; O = O2; }
  const bool do_sig = ((int)blockIdx.z == sigmoid_z);

  // double-buffered 128x64 bf16 tiles (144B padded rows): 2*2*18KB = 72KB LDS
  __shared__ __align__(16) unsigned int As[2][128 * LSTRIDE];
  __shared__ __align__(16) unsigned int Ws[2][128 * LSTRIDE];

  const int blockRow = blockIdx.y * 128;
  const int blockCol = blockIdx.x * 128;
  const int lane  = threadIdx.x & 31;
  const int wave  = threadIdx.x >> 5;      // 0..7
  const int waveN = wave & 3;              // 4 waves across N (32 cols each)
  const int waveM = wave >> 2;             // 2 waves across M (64 rows each)

  const v8f vzero = {0.f, 0.f, 0.f, 0.f, 0.f, 0.f, 0.f, 0.f};
  v8f acc[4][2];
  for (int mt = 0; mt < 4; ++mt)
    for (int nt = 0; nt < 2; ++nt)
      acc[mt][nt] = vzero;

  const uint4* gA = (const uint4*)A;   // 8 bf16 per uint4; row pitch = 2048/8 = 256
  const uint4* gW = (const uint4*)W;
  const int r    = threadIdx.x >> 1;   // tile row 0..127
  const int part = threadIdx.x & 1;    // half-row (64B) per thread

#if USE_ASYNC_LDS
  // ---- async prologue: stage tile 0 into buffer 0 (memory -> LDS direct)
  {
    const size_t aBase = (size_t)(blockRow + r) * (Cdim / 8) + part * 4;
    const size_t wBase = (size_t)(blockCol + r) * (Cdim / 8) + part * 4;
    unsigned int* dstA = &As[0][r * LSTRIDE + part * 16];
    unsigned int* dstW = &Ws[0][r * LSTRIDE + part * 16];
    #pragma unroll
    for (int q = 0; q < 4; ++q) {
      GL_ASYNC_B128(gA + aBase + q, dstA + q * 4);
      GL_ASYNC_B128(gW + wBase + q, dstW + q * 4);
    }
  }
  __builtin_amdgcn_s_wait_asynccnt(0);
  __syncthreads();

  int cur = 0;
  for (int kk = 0; kk < Cdim; kk += 64) {
    // ---- issue async copies for the next K tile into the other buffer
    if (kk + 64 < Cdim) {
      const size_t aBase = (size_t)(blockRow + r) * (Cdim / 8) + ((kk + 64) >> 3) + part * 4;
      const size_t wBase = (size_t)(blockCol + r) * (Cdim / 8) + ((kk + 64) >> 3) + part * 4;
      unsigned int* dstA = &As[cur ^ 1][r * LSTRIDE + part * 16];
      unsigned int* dstW = &Ws[cur ^ 1][r * LSTRIDE + part * 16];
      #pragma unroll
      for (int q = 0; q < 4; ++q) {
        GL_ASYNC_B128(gA + aBase + q, dstA + q * 4);
        GL_ASYNC_B128(gW + wBase + q, dstW + q * 4);
      }
    }
    // ---- 16 WMMAs per wave on the current buffer
    const unsigned int* Ab = As[cur];
    const unsigned int* Wb = Ws[cur];
    #pragma unroll
    for (int h2 = 0; h2 < 2; ++h2) {
      v16bf bf0 = load_bfrag(Wb, waveN * 2 + 0, lane, h2);
      v16bf bf1 = load_bfrag(Wb, waveN * 2 + 1, lane, h2);
      #pragma unroll
      for (int mt = 0; mt < 4; ++mt) {
        v16bf af = load_afrag(Ab, waveM * 4 + mt, lane, h2);
        acc[mt][0] = __builtin_amdgcn_wmma_f32_16x16x32_bf16(
            false, af, false, bf0, (short)0, acc[mt][0], false, false);
        acc[mt][1] = __builtin_amdgcn_wmma_f32_16x16x32_bf16(
            false, af, false, bf1, (short)0, acc[mt][1], false, false);
      }
    }
    __builtin_amdgcn_s_wait_asynccnt(0);
    __syncthreads();
    cur ^= 1;
  }
#else
  // ---- fallback: double-buffered reg-bounce staging
  uint4 ra[4], rw[4];
  {
    const size_t aBase = (size_t)(blockRow + r) * (Cdim / 8) + part * 4;
    const size_t wBase = (size_t)(blockCol + r) * (Cdim / 8) + part * 4;
    #pragma unroll
    for (int q = 0; q < 4; ++q) { ra[q] = gA[aBase + q]; rw[q] = gW[wBase + q]; }
    uint4* dA = (uint4*)&As[0][r * LSTRIDE + part * 16];
    uint4* dW = (uint4*)&Ws[0][r * LSTRIDE + part * 16];
    #pragma unroll
    for (int q = 0; q < 4; ++q) { dA[q] = ra[q]; dW[q] = rw[q]; }
  }
  __syncthreads();

  int cur = 0;
  for (int kk = 0; kk < Cdim; kk += 64) {
    const bool have_next = (kk + 64 < Cdim);
    if (have_next) {
      const size_t aBase = (size_t)(blockRow + r) * (Cdim / 8) + ((kk + 64) >> 3) + part * 4;
      const size_t wBase = (size_t)(blockCol + r) * (Cdim / 8) + ((kk + 64) >> 3) + part * 4;
      #pragma unroll
      for (int q = 0; q < 4; ++q) { ra[q] = gA[aBase + q]; rw[q] = gW[wBase + q]; }
      if (kk + 128 < Cdim) {
        __builtin_prefetch((const void*)(gA + aBase + 8), 0, 0);
        __builtin_prefetch((const void*)(gW + wBase + 8), 0, 0);
      }
    }
    const unsigned int* Ab = As[cur];
    const unsigned int* Wb = Ws[cur];
    #pragma unroll
    for (int h2 = 0; h2 < 2; ++h2) {
      v16bf bf0 = load_bfrag(Wb, waveN * 2 + 0, lane, h2);
      v16bf bf1 = load_bfrag(Wb, waveN * 2 + 1, lane, h2);
      #pragma unroll
      for (int mt = 0; mt < 4; ++mt) {
        v16bf af = load_afrag(Ab, waveM * 4 + mt, lane, h2);
        acc[mt][0] = __builtin_amdgcn_wmma_f32_16x16x32_bf16(
            false, af, false, bf0, (short)0, acc[mt][0], false, false);
        acc[mt][1] = __builtin_amdgcn_wmma_f32_16x16x32_bf16(
            false, af, false, bf1, (short)0, acc[mt][1], false, false);
      }
    }
    if (have_next) {
      uint4* dA = (uint4*)&As[cur ^ 1][r * LSTRIDE + part * 16];
      uint4* dW = (uint4*)&Ws[cur ^ 1][r * LSTRIDE + part * 16];
      #pragma unroll
      for (int q = 0; q < 4; ++q) { dA[q] = ra[q]; dW[q] = rw[q]; }
    }
    __syncthreads();
    cur ^= 1;
  }
#endif

  // ---- epilogue: C/D layout -> lane = col (N), VGPR idx + half*8 = row (M)
  const int colBase = blockCol + waveN * 32;
  const int rowBase = blockRow + waveM * 64 + (lane >> 4) * 8;
  #pragma unroll
  for (int mt = 0; mt < 4; ++mt) {
    #pragma unroll
    for (int nt = 0; nt < 2; ++nt) {
      const int col = colBase + nt * 16 + (lane & 15);
      const int row0 = rowBase + mt * 16;
      #pragma unroll
      for (int e = 0; e < 8; ++e) {
        float val = acc[mt][nt][e];
        if (do_sig) val = sigmoidf(val);
        O[(size_t)(row0 + e) * Cdim + col] = val;
      }
    }
  }
}

// ---------------------------------------------------------------------------
// 5) fp32 linear recurrence over T; emits rw = r * ((state*first + k) * v) bf16
// ---------------------------------------------------------------------------
__global__ void recur_kernel(const float* __restrict__ kbuf, const float* __restrict__ vbuf,
                             const float* __restrict__ rbuf,
                             const float* __restrict__ time_decay,
                             const float* __restrict__ time_first,
                             unsigned short* __restrict__ rw) {
  const int idx = blockIdx.x * blockDim.x + threadIdx.x;  // 0 .. B*C-1
  const int b = idx >> 11;
  const int c = idx & (Cdim - 1);
  const float decay = sigmoidf(time_decay[c]);
  const float first = __expf(time_first[c]);
  float state = 0.f;
  size_t off = (size_t)b * Tdim * Cdim + c;
  #pragma unroll 4
  for (int t = 0; t < Tdim; ++t) {
    const float kt = kbuf[off];
    const float vt = vbuf[off];
    const float rt = rbuf[off];
    state = state * decay + kt;
    const float rwkv = (state * first + kt) * vt;
    rw[off] = f2bf(rt * rwkv);
    off += Cdim;
  }
}

// ---------------------------------------------------------------------------
// launch
// ---------------------------------------------------------------------------
extern "C" void kernel_launch(void* const* d_in, const int* in_sizes, int n_in,
                              void* d_out, int out_size, void* d_ws, size_t ws_size,
                              hipStream_t stream) {
  const float* x    = (const float*)d_in[0];
  const float* ln_w = (const float*)d_in[1];
  const float* ln_b = (const float*)d_in[2];
  const float* mk   = (const float*)d_in[3];
  const float* mv   = (const float*)d_in[4];
  const float* mr   = (const float*)d_in[5];
  const float* td   = (const float*)d_in[6];
  const float* tf   = (const float*)d_in[7];
  const float* Wk   = (const float*)d_in[8];
  const float* Wv   = (const float*)d_in[9];
  const float* Wr   = (const float*)d_in[10];
  const float* Wo   = (const float*)d_in[11];
  float* out        = (float*)d_out;

  // workspace layout
  char* p = (char*)d_ws;
  float*          xt  = (float*)p;          p += NELEM * 4;
  unsigned short* xk  = (unsigned short*)p; p += NELEM * 2;
  unsigned short* xv  = (unsigned short*)p; p += NELEM * 2;
  unsigned short* xr  = (unsigned short*)p; p += NELEM * 2;
  float*          kb  = (float*)p;          p += NELEM * 4;
  float*          vb  = (float*)p;          p += NELEM * 4;
  float*          rb  = (float*)p;          p += NELEM * 4;
  unsigned short* rw  = (unsigned short*)p; p += NELEM * 2;
  unsigned short* wkb = (unsigned short*)p; p += (size_t)Cdim * Cdim * 2;
  unsigned short* wvb = (unsigned short*)p; p += (size_t)Cdim * Cdim * 2;
  unsigned short* wrb = (unsigned short*)p; p += (size_t)Cdim * Cdim * 2;
  unsigned short* wob = (unsigned short*)p; p += (size_t)Cdim * Cdim * 2;

  const size_t wElem = (size_t)Cdim * Cdim;       // 4,194,304
  const int wBlocks  = (int)(wElem / 1024);       // 4 floats per thread, 256 thr
  wconv_kernel<<<wBlocks, 256, 0, stream>>>(Wk, (unsigned int*)wkb);
  wconv_kernel<<<wBlocks, 256, 0, stream>>>(Wv, (unsigned int*)wvb);
  wconv_kernel<<<wBlocks, 256, 0, stream>>>(Wr, (unsigned int*)wrb);
  wconv_kernel<<<wBlocks, 256, 0, stream>>>(Wo, (unsigned int*)wob);

  ln_kernel<<<Mrows, 256, 0, stream>>>(x, ln_w, ln_b, xt);

  const int mixBlocks = (int)(NELEM / 4 / 256);   // 16384
  mix_kernel<<<mixBlocks, 256, 0, stream>>>(xt, mk, mv, mr,
                                            (unsigned int*)xk, (unsigned int*)xv,
                                            (unsigned int*)xr);

  dim3 ggrid(Cdim / 128, Mrows / 128, 3);         // (16, 64, 3)
  gemm_bf16_kernel<<<ggrid, 256, 0, stream>>>(xk, xv, xr, wkb, wvb, wrb,
                                              kb, vb, rb, /*sigmoid_z=*/2);

  recur_kernel<<<(Bdim * Cdim) / 256, 256, 0, stream>>>(kb, vb, rb, td, tf, rw);

  dim3 ogrid(Cdim / 128, Mrows / 128, 1);
  gemm_bf16_kernel<<<ogrid, 256, 0, stream>>>(rw, rw, rw, wob, wob, wob,
                                              out, out, out, /*sigmoid_z=*/-1);
}
